// TransformerModel_47296179863829
// MI455X (gfx1250) — matrix operand order repaired and checked by reading, not compiled, not run on hardware
//
#include <hip/hip_runtime.h>
#include <hip/hip_bf16.h>

// ---------------- types ----------------
typedef __attribute__((ext_vector_type(16))) __bf16        v16bf;
typedef __attribute__((ext_vector_type(8)))  float         v8f;
typedef __attribute__((ext_vector_type(4)))  unsigned int  u32x4;

union FragU { u32x4 q[2]; v16bf v; };   // 32 bytes

static constexpr int Bsz = 8, Ssz = 1024, Dsz = 512, Hn = 8, DH = 64;
static constexpr int Esz = 2048, Ln = 6, Msz = Bsz * Ssz;   // M = 8192

// ---------------- optional gfx1250 async/TDM paths (probe-guarded) ----------------
#if defined(__has_builtin)
#  if __has_builtin(__builtin_amdgcn_global_load_async_to_lds_b128) && \
      __has_builtin(__builtin_amdgcn_s_wait_asynccnt)
#    define USE_ASYNC_LDS 1
#  endif
#  if __has_builtin(__builtin_amdgcn_tensor_load_to_lds) && \
      __has_builtin(__builtin_amdgcn_s_wait_tensorcnt)
#    define HAVE_TDM 1
#  endif
#endif
#ifndef USE_ASYNC_LDS
#  define USE_ASYNC_LDS 0
#endif

#if USE_ASYNC_LDS
typedef int v4i __attribute__((vector_size(16)));
typedef __attribute__((address_space(1))) v4i v4i_as1;
typedef __attribute__((address_space(3))) v4i v4i_as3;
// 16-byte async global->LDS copy (per-lane addresses), tracked by ASYNCcnt
__device__ __forceinline__ void async_cp16(const void* gsrc, void* ldst) {
    __builtin_amdgcn_global_load_async_to_lds_b128(
        (v4i_as1*)(unsigned long long)(size_t)gsrc,
        (v4i_as3*)(unsigned int)(size_t)ldst, 0, 0);
}
#endif

// ---------------- helpers ----------------
__device__ __forceinline__ unsigned short f2bf(float f) {
    union { float f; unsigned int u; } v; v.f = f;
    unsigned int u = v.u + 0x7FFFu + ((v.u >> 16) & 1u);   // RNE
    return (unsigned short)(u >> 16);
}

__device__ __forceinline__ v8f wmma_bf16(v16bf a, v16bf b, v8f c) {
    return __builtin_amdgcn_wmma_f32_16x16x32_bf16(false, a, false, b,
                                                   (short)0, c, false, false);
}

// A-fragment (16xK=32, bf16): lane holds row (lane&15); K pattern:
//   i<8  -> k = i + half*8 ; i>=8 -> k = 16 + half*8 + (i-8)   (two b128 chunks)
__device__ __forceinline__ v16bf load_a_frag(const unsigned short* lds,
                                             int row0, int stride, int kbase) {
    int lane = threadIdx.x & 31;
    int hv = lane >> 4;
    const unsigned short* p = lds + (row0 + (lane & 15)) * stride + kbase + hv * 8;
    FragU f;
    f.q[0] = *(const u32x4*)(p);
    f.q[1] = *(const u32x4*)(p + 16);
    return f.v;
}

// B-fragment (K=32 x 16, bf16): lane holds column (lane&15); K pattern same as A.
__device__ __forceinline__ v16bf load_b_frag(const unsigned short* lds,
                                             int krow0, int stride, int nbase) {
    int lane = threadIdx.x & 31;
    int hv = lane >> 4;
    const unsigned short* p = lds + (krow0 + hv * 8) * stride + nbase + (lane & 15);
    union { unsigned short s[16]; v16bf v; } f;
#pragma unroll
    for (int i = 0; i < 8; ++i) {
        f.s[i]     = p[i * stride];
        f.s[i + 8] = p[(16 + i) * stride];
    }
    return f.v;
}

// ---------------- f32 -> bf16 convert ----------------
__global__ void conv_bf16_kernel(const float* __restrict__ in,
                                 unsigned short* __restrict__ out, int n) {
    int i = blockIdx.x * blockDim.x + threadIdx.x;
    if (i < n) out[i] = f2bf(in[i]);
}

// ---------------- LayerNorm (f32 -> bf16), one row (D=512) per block ----------------
__global__ __launch_bounds__(256)
void ln_kernel(const float* __restrict__ x, const float* __restrict__ g,
               const float* __restrict__ be, unsigned short* __restrict__ out) {
    __shared__ float red[256];
    int row = blockIdx.x, t = threadIdx.x;
    const float* xr = x + (size_t)row * Dsz;
    float v0 = xr[t], v1 = xr[t + 256];
    red[t] = v0 + v1;
    __syncthreads();
    for (int s = 128; s > 0; s >>= 1) {
        if (t < s) red[t] += red[t + s];
        __syncthreads();
    }
    float mu = red[0] * (1.0f / 512.0f);
    __syncthreads();
    float d0 = v0 - mu, d1 = v1 - mu;
    red[t] = d0 * d0 + d1 * d1;
    __syncthreads();
    for (int s = 128; s > 0; s >>= 1) {
        if (t < s) red[t] += red[t + s];
        __syncthreads();
    }
    float rs = rsqrtf(red[0] * (1.0f / 512.0f) + 1e-3f);
    out[(size_t)row * Dsz + t]       = f2bf(d0 * rs * g[t] + be[t]);
    out[(size_t)row * Dsz + t + 256] = f2bf(d1 * rs * g[t + 256] + be[t + 256]);
}

// ---------------- bf16 GEMM: C = act(A[MxK] * B[KxN] + bias) (+res) ----------------
// block tile 128x128, 8 waves as 2(M)x4(N), wave tile 64x32 (4x2 frags, 8 WMMA/kstep)
// double-buffered LDS; global->LDS via ASYNCcnt path when available.
template <bool OUT_BF16, bool RELU, bool HAS_BIAS, bool HAS_RES>
__global__ __launch_bounds__(256)
void gemm_bf16_kernel(const unsigned short* __restrict__ A,
                      const unsigned short* __restrict__ Bw,
                      const float* __restrict__ bias,
                      const float* __restrict__ res,
                      void* __restrict__ out, int N, int K) {
    __shared__ alignas(16) unsigned short As[2][128 * 32];
    __shared__ alignas(16) unsigned short Bs[2][32 * 128];

    const int t = threadIdx.x;
    const int tileN = blockIdx.x * 128;
    const int tileM = blockIdx.y * 128;
    const int wave = t >> 5, lane = t & 31;
    const int wm = wave >> 2, wn = wave & 3;     // 2 x 4 wave grid
    const int hv = lane >> 4, nl = lane & 15;

    v8f vz = {0.f, 0.f, 0.f, 0.f, 0.f, 0.f, 0.f, 0.f};
    v8f acc[4][2];
#pragma unroll
    for (int mt = 0; mt < 4; ++mt) { acc[mt][0] = vz; acc[mt][1] = vz; }

    // per-thread copy slots: A tile 128x32 (16 elems), B tile 32x128 (16 elems)
    const int ar = t >> 1, ah = (t & 1) * 16;
    const int br = t >> 3, bc = (t & 7) * 16;
    const unsigned short* gA = A + (size_t)(tileM + ar) * K + ah;
    const unsigned short* gB = Bw + (size_t)br * N + tileN + bc;
    const int nk = K >> 5;

#if USE_ASYNC_LDS
    // prologue: tile 0
    async_cp16(gA, &As[0][ar * 32 + ah]);
    async_cp16(gA + 8, &As[0][ar * 32 + ah + 8]);
    async_cp16(gB, &Bs[0][br * 128 + bc]);
    async_cp16(gB + 8, &Bs[0][br * 128 + bc + 8]);
#else
    u32x4 ra0, ra1, rb0, rb1;
    ra0 = *(const u32x4*)(gA);
    ra1 = *(const u32x4*)(gA + 8);
    rb0 = *(const u32x4*)(gB);
    rb1 = *(const u32x4*)(gB + 8);
    *(u32x4*)(&As[0][ar * 32 + ah])     = ra0;
    *(u32x4*)(&As[0][ar * 32 + ah + 8]) = ra1;
    *(u32x4*)(&Bs[0][br * 128 + bc])     = rb0;
    *(u32x4*)(&Bs[0][br * 128 + bc + 8]) = rb1;
#endif

    for (int kt = 0; kt < nk; ++kt) {
        const int cur = kt & 1;
        const int nxt = cur ^ 1;

        if (kt + 2 < nk) {   // L2 warm-up two tiles ahead
            __builtin_prefetch(gA + (size_t)(kt + 2) * 32, 0, 0);
            __builtin_prefetch(gB + (size_t)(kt + 2) * 32 * N, 0, 0);
        }

#if USE_ASYNC_LDS
        __builtin_amdgcn_s_wait_asynccnt(0);     // tile kt landed in LDS[cur]
        __syncthreads();
        if (kt + 1 < nk) {                       // kick tile kt+1 during compute
            const unsigned short* a1p = gA + (size_t)(kt + 1) * 32;
            const unsigned short* b1p = gB + (size_t)(kt + 1) * 32 * N;
            async_cp16(a1p, &As[nxt][ar * 32 + ah]);
            async_cp16(a1p + 8, &As[nxt][ar * 32 + ah + 8]);
            async_cp16(b1p, &Bs[nxt][br * 128 + bc]);
            async_cp16(b1p + 8, &Bs[nxt][br * 128 + bc + 8]);
        }
#else
        if (kt + 1 < nk) {                       // issue global loads early
            const unsigned short* a1p = gA + (size_t)(kt + 1) * 32;
            const unsigned short* b1p = gB + (size_t)(kt + 1) * 32 * N;
            ra0 = *(const u32x4*)(a1p);
            ra1 = *(const u32x4*)(a1p + 8);
            rb0 = *(const u32x4*)(b1p);
            rb1 = *(const u32x4*)(b1p + 8);
        }
        __syncthreads();                         // LDS[cur] stores visible
#endif

        const unsigned short* Ab = As[cur];
        const unsigned short* Bb = Bs[cur];
        v16bf af[4];
#pragma unroll
        for (int mt = 0; mt < 4; ++mt)
            af[mt] = load_a_frag(Ab, wm * 64 + mt * 16, 32, 0);
        v16bf b0f = load_b_frag(Bb, 0, 128, wn * 32);
        v16bf b1f = load_b_frag(Bb, 0, 128, wn * 32 + 16);
#pragma unroll
        for (int mt = 0; mt < 4; ++mt) {
            acc[mt][0] = wmma_bf16(af[mt], b0f, acc[mt][0]);
            acc[mt][1] = wmma_bf16(af[mt], b1f, acc[mt][1]);
        }

#if !USE_ASYNC_LDS
        if (kt + 1 < nk) {                       // commit next tile (other buffer)
            *(u32x4*)(&As[nxt][ar * 32 + ah])     = ra0;
            *(u32x4*)(&As[nxt][ar * 32 + ah + 8]) = ra1;
            *(u32x4*)(&Bs[nxt][br * 128 + bc])     = rb0;
            *(u32x4*)(&Bs[nxt][br * 128 + bc + 8]) = rb1;
        }
#endif
    }

#pragma unroll
    for (int mt = 0; mt < 4; ++mt)
#pragma unroll
        for (int nt = 0; nt < 2; ++nt) {
            int col = tileN + wn * 32 + nt * 16 + nl;
            float bv = HAS_BIAS ? bias[col] : 0.0f;
#pragma unroll
            for (int r = 0; r < 8; ++r) {
                int row = tileM + wm * 64 + mt * 16 + r + 8 * hv;
                float v = acc[mt][nt][r] + bv;
                if (RELU) v = fmaxf(v, 0.0f);
                size_t idx = (size_t)row * N + col;
                if (HAS_RES) v += res[idx];
                if (OUT_BF16) ((unsigned short*)out)[idx] = f2bf(v);
                else          ((float*)out)[idx] = v;
            }
        }
}

// ---------------- flash attention: one block per (batch, head, 64-row q tile) ----------------
__global__ __launch_bounds__(128)
void attn_kernel(const unsigned short* __restrict__ Q,
                 const unsigned short* __restrict__ K,
                 const unsigned short* __restrict__ V,
                 unsigned short* __restrict__ Z) {
    __shared__ alignas(16) unsigned short Qs[64 * 64];
    __shared__ alignas(16) unsigned short KsT[64 * 64];   // transposed: [d][key]
    __shared__ alignas(16) unsigned short Vs[64 * 64];    // [key][d]
    __shared__ alignas(16) unsigned short Ps[4][16 * 64]; // per-wave probs

    const int bid = blockIdx.x;
    const int qt = bid & 15, h = (bid >> 4) & 7, bb = bid >> 7;
    const int qbase = qt * 64;
    const int t = threadIdx.x, wave = t >> 5, lane = t & 31;
    const int hv = lane >> 4, nl = lane & 15;

    for (int idx = t; idx < 64 * 64; idx += 128) {
        int r = idx >> 6, c = idx & 63;
        Qs[idx] = Q[((size_t)(bb * Ssz + qbase + r)) * Dsz + h * DH + c];
    }
    __syncthreads();

    const v16bf qa0 = load_a_frag(Qs, wave * 16, 64, 0);
    const v16bf qa1 = load_a_frag(Qs, wave * 16, 64, 32);

    v8f vz = {0.f, 0.f, 0.f, 0.f, 0.f, 0.f, 0.f, 0.f};
    v8f oacc[4];
    oacc[0] = vz; oacc[1] = vz; oacc[2] = vz; oacc[3] = vz;
    float m_run[8], l_run[8];
#pragma unroll
    for (int r = 0; r < 8; ++r) { m_run[r] = -1e30f; l_run[r] = 0.0f; }

    unsigned short* myP = Ps[wave];

    for (int kt = 0; kt < Ssz / 64; ++kt) {
        const int kb = kt * 64;
        for (int idx = t; idx < 64 * 64; idx += 128) {
            int r = idx >> 6, c = idx & 63;
            size_t gsrc = ((size_t)(bb * Ssz + kb + r)) * Dsz + h * DH + c;
            KsT[c * 64 + r] = K[gsrc];
            Vs[idx] = V[gsrc];
        }
        __syncthreads();

        // scores: 16 q-rows x 64 keys per wave
        v8f sf[4];
#pragma unroll
        for (int nt = 0; nt < 4; ++nt) {
            v8f s = vz;
            s = wmma_bf16(qa0, load_b_frag(KsT, 0,  64, nt * 16), s);
            s = wmma_bf16(qa1, load_b_frag(KsT, 32, 64, nt * 16), s);
            sf[nt] = s * 0.125f;                 // 1/sqrt(DH)
        }

        // online softmax per C-row (row m = r + 8*half lives in one 16-lane half)
#pragma unroll
        for (int r = 0; r < 8; ++r) {
            float mx = fmaxf(fmaxf(sf[0][r], sf[1][r]), fmaxf(sf[2][r], sf[3][r]));
#pragma unroll
            for (int off = 1; off <= 8; off <<= 1)
                mx = fmaxf(mx, __shfl_xor(mx, off, 32));
            float newm = fmaxf(m_run[r], mx);
            float corr = __expf(m_run[r] - newm);
            float rowsum = 0.0f;
#pragma unroll
            for (int nt = 0; nt < 4; ++nt) {
                float p = __expf(sf[nt][r] - newm);
                rowsum += p;
                myP[(r + 8 * hv) * 64 + nt * 16 + nl] = f2bf(p);
            }
#pragma unroll
            for (int off = 1; off <= 8; off <<= 1)
                rowsum += __shfl_xor(rowsum, off, 32);
            m_run[r] = newm;
            l_run[r] = l_run[r] * corr + rowsum;
#pragma unroll
            for (int dt = 0; dt < 4; ++dt) oacc[dt][r] *= corr;
        }

        // O += P * V   (K-dim = 64 keys, two 32-key WMMA steps)
#pragma unroll
        for (int dt = 0; dt < 4; ++dt) {
#pragma unroll
            for (int ks = 0; ks < 2; ++ks) {
                v16bf pa = load_a_frag(myP, 0, 64, ks * 32);
                v16bf vb = load_b_frag(Vs, ks * 32, 64, dt * 16);
                oacc[dt] = wmma_bf16(pa, vb, oacc[dt]);
            }
        }
        __syncthreads();
    }

#pragma unroll
    for (int dt = 0; dt < 4; ++dt)
#pragma unroll
        for (int r = 0; r < 8; ++r) {
            int row = bb * Ssz + qbase + wave * 16 + r + 8 * hv;
            Z[(size_t)row * Dsz + h * DH + dt * 16 + nl] =
                f2bf(oacc[dt][r] / l_run[r]);
        }
}

// ---------------- TDM probe (unlaunched; checks tensor_load_to_lds lowering) ----------------
#if defined(HAVE_TDM)
typedef __attribute__((ext_vector_type(8))) int i32x8;
typedef __attribute__((ext_vector_type(4))) int i32x4;
__global__ void tdm_probe_kernel(float* out) {
    u32x4 g0 = {0u, 0u, 0u, 0u};
    i32x8 g1 = {0, 0, 0, 0, 0, 0, 0, 0};
    i32x4 g2 = {0, 0, 0, 0};
    i32x4 g3 = {0, 0, 0, 0};
    i32x8 g4 = {0, 0, 0, 0, 0, 0, 0, 0};
    __builtin_amdgcn_tensor_load_to_lds(g0, g1, g2, g3, g4, 0);
    __builtin_amdgcn_s_wait_tensorcnt(0);
    if (out) out[0] = 0.f;
}
#endif

// ---------------- host launch ----------------
extern "C" void kernel_launch(void* const* d_in, const int* in_sizes, int n_in,
                              void* d_out, int out_size, void* d_ws, size_t ws_size,
                              hipStream_t stream) {
    (void)in_sizes; (void)n_in; (void)out_size; (void)ws_size;

    const float* x      = (const float*)d_in[0];
    const float* Wq     = (const float*)d_in[1];
    const float* Wk     = (const float*)d_in[2];
    const float* Wv     = (const float*)d_in[3];
    const float* Wo     = (const float*)d_in[4];
    const float* W1     = (const float*)d_in[5];
    const float* b1     = (const float*)d_in[6];
    const float* W2     = (const float*)d_in[7];
    const float* b2     = (const float*)d_in[8];
    const float* g_attn = (const float*)d_in[9];
    const float* be_attn= (const float*)d_in[10];
    const float* g_ffn  = (const float*)d_in[11];
    const float* be_ffn = (const float*)d_in[12];

    float* enc = (float*)d_out;   // running residual stream (B*S*D f32)

    // workspace partition
    char* p = (char*)d_ws;
    auto alloc = [&](size_t bytes) {
        char* r = p;
        p += (bytes + 255) & ~(size_t)255;
        return r;
    };
    const size_t nDD = (size_t)Ln * Dsz * Dsz;
    const size_t nDE = (size_t)Ln * Dsz * Esz;
    unsigned short* wq_bf = (unsigned short*)alloc(nDD * 2);
    unsigned short* wk_bf = (unsigned short*)alloc(nDD * 2);
    unsigned short* wv_bf = (unsigned short*)alloc(nDD * 2);
    unsigned short* wo_bf = (unsigned short*)alloc(nDD * 2);
    unsigned short* w1_bf = (unsigned short*)alloc(nDE * 2);
    unsigned short* w2_bf = (unsigned short*)alloc(nDE * 2);
    const size_t nAct = (size_t)Msz * Dsz;
    unsigned short* ln_bf = (unsigned short*)alloc(nAct * 2);
    unsigned short* q_bf  = (unsigned short*)alloc(nAct * 2);
    unsigned short* k_bf  = (unsigned short*)alloc(nAct * 2);
    unsigned short* v_bf  = (unsigned short*)alloc(nAct * 2);
    unsigned short* z_bf  = (unsigned short*)alloc(nAct * 2);
    unsigned short* h_bf  = (unsigned short*)alloc((size_t)Msz * Esz * 2);

    auto conv = [&](const float* src, unsigned short* dst, size_t n) {
        conv_bf16_kernel<<<dim3((unsigned)((n + 255) / 256)), 256, 0, stream>>>(
            src, dst, (int)n);
    };
    conv(Wq, wq_bf, nDD); conv(Wk, wk_bf, nDD); conv(Wv, wv_bf, nDD);
    conv(Wo, wo_bf, nDD); conv(W1, w1_bf, nDE); conv(W2, w2_bf, nDE);

    (void)hipMemcpyAsync(enc, x, (size_t)Msz * Dsz * sizeof(float),
                         hipMemcpyDeviceToDevice, stream);

    const dim3 blk(256);
    const dim3 gridD(Dsz / 128, Msz / 128);   // N=512
    const dim3 gridE(Esz / 128, Msz / 128);   // N=2048

    for (int l = 0; l < Ln; ++l) {
        const unsigned short* wql = wq_bf + (size_t)l * Dsz * Dsz;
        const unsigned short* wkl = wk_bf + (size_t)l * Dsz * Dsz;
        const unsigned short* wvl = wv_bf + (size_t)l * Dsz * Dsz;
        const unsigned short* wol = wo_bf + (size_t)l * Dsz * Dsz;
        const unsigned short* w1l = w1_bf + (size_t)l * Dsz * Esz;
        const unsigned short* w2l = w2_bf + (size_t)l * Esz * Dsz;

        // --- attention block ---
        ln_kernel<<<Msz, blk, 0, stream>>>(enc, g_attn + l * Dsz,
                                           be_attn + l * Dsz, ln_bf);
        gemm_bf16_kernel<true, false, false, false><<<gridD, blk, 0, stream>>>(
            ln_bf, wql, nullptr, nullptr, q_bf, Dsz, Dsz);
        gemm_bf16_kernel<true, false, false, false><<<gridD, blk, 0, stream>>>(
            ln_bf, wkl, nullptr, nullptr, k_bf, Dsz, Dsz);
        gemm_bf16_kernel<true, false, false, false><<<gridD, blk, 0, stream>>>(
            ln_bf, wvl, nullptr, nullptr, v_bf, Dsz, Dsz);
        attn_kernel<<<Bsz * Hn * (Ssz / 64), 128, 0, stream>>>(q_bf, k_bf, v_bf,
                                                               z_bf);
        // encoded += z @ Wo   (in-place residual, f32 out)
        gemm_bf16_kernel<false, false, false, true><<<gridD, blk, 0, stream>>>(
            z_bf, wol, nullptr, enc, enc, Dsz, Dsz);

        // --- FFN block ---
        ln_kernel<<<Msz, blk, 0, stream>>>(enc, g_ffn + l * Dsz,
                                           be_ffn + l * Dsz, ln_bf);
        gemm_bf16_kernel<true, true, true, false><<<gridE, blk, 0, stream>>>(
            ln_bf, w1l, b1 + (size_t)l * Esz, nullptr, h_bf, Esz, Dsz);
        gemm_bf16_kernel<false, false, true, true><<<gridD, blk, 0, stream>>>(
            h_bf, w2l, b2 + (size_t)l * Dsz, enc, enc, Dsz, Esz);
    }
}
